// GPMultiLineRecognitionREDecoder_58213986730274
// MI455X (gfx1250) — matrix operand order repaired and checked by reading, not compiled, not run on hardware
//
#include <hip/hip_runtime.h>
#include <hip/hip_bf16.h>
#include <stdint.h>

// Problem constants (from reference)
#define HIDDEN    992
#define LABEL_EMB 32
#define TOTAL     1024
#define HEAD_SIZE 64
#define OUT_DIM   128           // 2*HEAD_SIZE
#define NW        3             // head / tail / t2h
#define NB        4
#define NS        2048
#define M_TOTAL   (NB * NS)     // 8192
#define SCALE_F   0.125f        // 1/sqrt(64)
#define INF_F     1000000000000.0f

typedef __attribute__((ext_vector_type(16))) __bf16 v16bf;
typedef __attribute__((ext_vector_type(8)))  __bf16 v8bf;
typedef __attribute__((ext_vector_type(8)))  float  v8f;

__device__ __forceinline__ unsigned short f2bf(float f) {
    unsigned int u = __float_as_uint(f);
    u += 0x7FFFu + ((u >> 16) & 1u);   // round-to-nearest-even
    return (unsigned short)(u >> 16);
}

// gfx1250 async global->LDS copy, 16B per lane. VDST operand carries the LDS
// byte address (low 32 bits of the generic pointer to __shared__ memory);
// VADDR is the 64-bit global address. Tracked by ASYNCcnt.
__device__ __forceinline__ void async_copy_b128(unsigned int lds_off, const void* gptr) {
    asm volatile("global_load_async_to_lds_b128 %0, %1, off"
                 :: "v"(lds_off), "v"(gptr)
                 : "memory");
}
__device__ __forceinline__ void wait_asynccnt0() {
    asm volatile("s_wait_asynccnt 0x0" ::: "memory");
}

// A fragment (16x32 bf16, MxK). ISA layout: lane holds row M=lane%16;
// lanes 0-15: K 0..7 then 16..23 ; lanes 16-31: K 8..15 then 24..31.
__device__ __forceinline__ v16bf load_a_frag(const unsigned short* base, int stride,
                                             int row0, int k0) {
    int lane = threadIdx.x & 31;
    int row  = row0 + (lane & 15);
    int kh   = (lane >> 4) * 8;
    v8bf lo = *(const v8bf*)(base + row * stride + k0 + kh);
    v8bf hi = *(const v8bf*)(base + row * stride + k0 + 16 + kh);
    return __builtin_shufflevector(lo, hi, 0,1,2,3,4,5,6,7,8,9,10,11,12,13,14,15);
}

// B fragment (32x16 bf16, KxN) read from an [N][K] row-major buffer.
// ISA layout: lane holds column N=lane%16; lanes 0-15: K 0..7 then 16..23 ; lanes 16-31: K 8..15 then 24..31.
__device__ __forceinline__ v16bf load_b_frag(const unsigned short* base, int stride,
                                             int n0, int k0) {
    int lane = threadIdx.x & 31;
    int n    = n0 + (lane & 15);
    int kh   = (lane >> 4) * 8;
    v8bf lo = *(const v8bf*)(base + n * stride + k0 + kh);
    v8bf hi = *(const v8bf*)(base + n * stride + k0 + 16 + kh);
    return __builtin_shufflevector(lo, hi, 0,1,2,3,4,5,6,7,8,9,10,11,12,13,14,15);
}

// ---------------------------------------------------------------------------
// Kernel 1: fused concat + bf16 cast + projection GEMM.
//   qk[w][m][0:64]  = ( x[m]·W_w^T[:, :64] + b_w[:64] ) * SCALE   (q, scale folded)
//   qk[w][m][64:128]=   x[m]·W_w^T[:,64:] + b_w[64:]              (k)
// M=8192, N=128 per weight set, K=1024. Block tile 64(M)x128(N), K step 32.
// (Staging stays explicit here: data must be converted f32->bf16 en route.)
// ---------------------------------------------------------------------------
__global__ __launch_bounds__(256)
void proj_kernel(const float* __restrict__ hidden,
                 const int*   __restrict__ labels,
                 const float* __restrict__ emb,
                 const float* __restrict__ W0, const float* __restrict__ W1,
                 const float* __restrict__ W2,
                 const float* __restrict__ bb0, const float* __restrict__ bb1,
                 const float* __restrict__ bb2,
                 unsigned short* __restrict__ qk) {
    __shared__ unsigned short As[64 * 32];    // [m][k] bf16 tile
    __shared__ unsigned short Bs[128 * 32];   // [n][k] bf16 tile (W rows are [n][k])

    const int w = blockIdx.y;
    const float* W    = (w == 0) ? W0  : (w == 1) ? W1  : W2;
    const float* bias = (w == 0) ? bb0 : (w == 1) ? bb1 : bb2;

    const int m0   = blockIdx.x * 64;
    const int t    = threadIdx.x;
    const int wave = t >> 5;
    const int wm   = wave & 1;       // 2 M-subtiles of 32
    const int wn   = wave >> 1;      // 4 N-subtiles of 32

    v8f acc[2][2] = {};

    // Per-thread load assignments
    const int arow = t >> 2;               // 0..63
    const int achk = (t & 3) * 8;          // 0,8,16,24
    const int mg   = m0 + arow;
    const int lab  = labels[mg];           // entity label for this row
    const int brow = t >> 1;               // 0..127 (N index)
    const int bchk = (t & 1) * 16;         // 0,16

    for (int kt = 0; kt < TOTAL / 32; ++kt) {
        // ---- A tile: fused concat(hidden, emb[label]) with f32->bf16 cast
        {
            int col = kt * 32 + achk;      // 8-aligned; never straddles 992
            const float* asrc = (col < HIDDEN)
                ? (hidden + (size_t)mg * HIDDEN + col)
                : (emb + (size_t)lab * LABEL_EMB + (col - HIDDEN));
            float4 a0 = ((const float4*)asrc)[0];
            float4 a1 = ((const float4*)asrc)[1];
            unsigned short* ad = As + arow * 32 + achk;
            ad[0] = f2bf(a0.x); ad[1] = f2bf(a0.y); ad[2] = f2bf(a0.z); ad[3] = f2bf(a0.w);
            ad[4] = f2bf(a1.x); ad[5] = f2bf(a1.y); ad[6] = f2bf(a1.z); ad[7] = f2bf(a1.w);
            if (col + 32 < HIDDEN)
                __builtin_prefetch(asrc + 32, 0, 1);   // next K tile -> global_prefetch
        }
        // ---- B tile: W rows (already [n][k]) with f32->bf16 cast
        {
            int bcol = kt * 32 + bchk;
            const float* bsrc = W + (size_t)brow * TOTAL + bcol;
            float4 b0 = ((const float4*)bsrc)[0];
            float4 b1 = ((const float4*)bsrc)[1];
            float4 b2 = ((const float4*)bsrc)[2];
            float4 b3 = ((const float4*)bsrc)[3];
            unsigned short* bd = Bs + brow * 32 + bchk;
            bd[ 0]=f2bf(b0.x); bd[ 1]=f2bf(b0.y); bd[ 2]=f2bf(b0.z); bd[ 3]=f2bf(b0.w);
            bd[ 4]=f2bf(b1.x); bd[ 5]=f2bf(b1.y); bd[ 6]=f2bf(b1.z); bd[ 7]=f2bf(b1.w);
            bd[ 8]=f2bf(b2.x); bd[ 9]=f2bf(b2.y); bd[10]=f2bf(b2.z); bd[11]=f2bf(b2.w);
            bd[12]=f2bf(b3.x); bd[13]=f2bf(b3.y); bd[14]=f2bf(b3.z); bd[15]=f2bf(b3.w);
            if (kt + 1 < TOTAL / 32)
                __builtin_prefetch(bsrc + 32, 0, 1);
        }
        __syncthreads();

        v16bf afr0 = load_a_frag(As, 32, wm * 32 +  0, 0);
        v16bf afr1 = load_a_frag(As, 32, wm * 32 + 16, 0);
        v16bf bfr0 = load_b_frag(Bs, 32, wn * 32 +  0, 0);
        v16bf bfr1 = load_b_frag(Bs, 32, wn * 32 + 16, 0);

        acc[0][0] = __builtin_amdgcn_wmma_f32_16x16x32_bf16(false, afr0, false, bfr0,
                                                            (short)0, acc[0][0], false, false);
        acc[0][1] = __builtin_amdgcn_wmma_f32_16x16x32_bf16(false, afr0, false, bfr1,
                                                            (short)0, acc[0][1], false, false);
        acc[1][0] = __builtin_amdgcn_wmma_f32_16x16x32_bf16(false, afr1, false, bfr0,
                                                            (short)0, acc[1][0], false, false);
        acc[1][1] = __builtin_amdgcn_wmma_f32_16x16x32_bf16(false, afr1, false, bfr1,
                                                            (short)0, acc[1][1], false, false);
        __syncthreads();
    }

    // ---- epilogue: add bias, fold SCALE into q half, cast to bf16, store
    const int lane = t & 31;
    const int rsel = (lane >> 4) * 8;   // C/D layout: lanes 16-31 hold M+8
    const int ncol = lane & 15;
    for (int mi = 0; mi < 2; ++mi) {
        for (int ni = 0; ni < 2; ++ni) {
            int   nloc = wn * 32 + ni * 16 + ncol;        // 0..127
            float bval = bias[nloc];
            float sc   = (nloc < HEAD_SIZE) ? SCALE_F : 1.0f;
            for (int r = 0; r < 8; ++r) {
                int m = m0 + wm * 32 + mi * 16 + r + rsel;
                float v = (acc[mi][ni][r] + bval) * sc;
                qk[((size_t)w * M_TOTAL + m) * OUT_DIM + nloc] = f2bf(v);
            }
        }
    }
}

// ---------------------------------------------------------------------------
// Kernel 2: logits[b][w][m][n] = q[m]·k[n] (SCALE pre-folded) with mask.
// Per (b,w): M=N=2048, K=64. Block tile 128x128, K cols 0..15,32..47 staged in LDS once
// via gfx1250 async global->LDS copies (ASYNCcnt path, no VGPR round-trip).
// ---------------------------------------------------------------------------
__global__ __launch_bounds__(256)
void logits_kernel(const unsigned short* __restrict__ qk,
                   const float* __restrict__ amask,
                   float* __restrict__ out) {
    __shared__ unsigned short Qs[128 * 64];
    __shared__ unsigned short Ks[128 * 64];

    const int z  = blockIdx.z;           // b*NW + w  (matches output layout)
    const int b  = z / NW;
    const int w  = z % NW;
    const int m0 = blockIdx.x * 128;
    const int n0 = blockIdx.y * 128;
    const unsigned short* base = qk + ((size_t)w * M_TOTAL + (size_t)b * NS) * OUT_DIM;

    const int t = threadIdx.x;
    {   // stage q rows [m0,m0+128) cols 0..15,32..47 and k rows [n0,n0+128) cols 64..79,96..111
        // Each thread moves 2x16B for Q and 2x16B for K via async-to-LDS.
        int row  = t >> 1;
        int half = (t & 1) * 32;         // copies 16 bf16 = 32 B per thread per tile
        const unsigned short* qsrc = base + (size_t)(m0 + row) * OUT_DIM + half;
        const unsigned short* ksrc = base + (size_t)(n0 + row) * OUT_DIM + 64 + half;
        unsigned int qdst = (unsigned int)(uintptr_t)(Qs + row * 64 + half);
        unsigned int kdst = (unsigned int)(uintptr_t)(Ks + row * 64 + half);
        async_copy_b128(qdst,      qsrc);
        async_copy_b128(qdst + 16, qsrc + 8);
        async_copy_b128(kdst,      ksrc);
        async_copy_b128(kdst + 16, ksrc + 8);
        wait_asynccnt0();
    }
    __syncthreads();

    const int wave = t >> 5;
    const int wm   = wave & 3;           // 4 M-subtiles of 32
    const int wn   = wave >> 2;          // 2 N-subtiles of 64

    v8f acc[2][4] = {};
    for (int kk = 0; kk < HEAD_SIZE; kk += 32) {
        v16bf a0 = load_a_frag(Qs, 64, wm * 32 +  0, kk);
        v16bf a1 = load_a_frag(Qs, 64, wm * 32 + 16, kk);
        v16bf bf0 = load_b_frag(Ks, 64, wn * 64 +  0, kk);
        v16bf bf1 = load_b_frag(Ks, 64, wn * 64 + 16, kk);
        v16bf bf2 = load_b_frag(Ks, 64, wn * 64 + 32, kk);
        v16bf bf3 = load_b_frag(Ks, 64, wn * 64 + 48, kk);

        acc[0][0] = __builtin_amdgcn_wmma_f32_16x16x32_bf16(false, a0, false, bf0, (short)0, acc[0][0], false, false);
        acc[0][1] = __builtin_amdgcn_wmma_f32_16x16x32_bf16(false, a0, false, bf1, (short)0, acc[0][1], false, false);
        acc[0][2] = __builtin_amdgcn_wmma_f32_16x16x32_bf16(false, a0, false, bf2, (short)0, acc[0][2], false, false);
        acc[0][3] = __builtin_amdgcn_wmma_f32_16x16x32_bf16(false, a0, false, bf3, (short)0, acc[0][3], false, false);
        acc[1][0] = __builtin_amdgcn_wmma_f32_16x16x32_bf16(false, a1, false, bf0, (short)0, acc[1][0], false, false);
        acc[1][1] = __builtin_amdgcn_wmma_f32_16x16x32_bf16(false, a1, false, bf1, (short)0, acc[1][1], false, false);
        acc[1][2] = __builtin_amdgcn_wmma_f32_16x16x32_bf16(false, a1, false, bf2, (short)0, acc[1][2], false, false);
        acc[1][3] = __builtin_amdgcn_wmma_f32_16x16x32_bf16(false, a1, false, bf3, (short)0, acc[1][3], false, false);
    }

    // ---- epilogue: mask + fp32 store (output-bandwidth bound part)
    const int lane = t & 31;
    const int rsel = (lane >> 4) * 8;
    const int ncol = lane & 15;
    for (int ni = 0; ni < 4; ++ni) {
        int   n   = n0 + wn * 64 + ni * 16 + ncol;
        float pad = amask[(size_t)b * NS + n];
        float neg = (1.0f - pad) * INF_F;
        for (int mi = 0; mi < 2; ++mi) {
            for (int r = 0; r < 8; ++r) {
                int m = m0 + wm * 32 + mi * 16 + r + rsel;
                out[((size_t)z * NS + m) * NS + n] = acc[mi][ni][r] * pad - neg;
            }
        }
    }
}

extern "C" void kernel_launch(void* const* d_in, const int* in_sizes, int n_in,
                              void* d_out, int out_size, void* d_ws, size_t ws_size,
                              hipStream_t stream) {
    (void)in_sizes; (void)n_in; (void)out_size; (void)ws_size;
    const float* hidden = (const float*)d_in[0];
    const int*   labels = (const int*)  d_in[1];
    const float* amask  = (const float*)d_in[2];
    const float* emb    = (const float*)d_in[3];
    const float* W0     = (const float*)d_in[4];
    const float* b0     = (const float*)d_in[5];
    const float* W1     = (const float*)d_in[6];
    const float* b1     = (const float*)d_in[7];
    const float* W2     = (const float*)d_in[8];
    const float* b2     = (const float*)d_in[9];

    unsigned short* qk = (unsigned short*)d_ws;   // [3][8192][128] bf16 = 6 MB

    dim3 gp(M_TOTAL / 64, NW, 1);                 // 128 x 3 blocks
    proj_kernel<<<gp, 256, 0, stream>>>(hidden, labels, emb, W0, W1, W2, b0, b1, b2, qk);

    dim3 gl(NS / 128, NS / 128, NB * NW);         // 16 x 16 x 12 blocks
    logits_kernel<<<gl, 256, 0, stream>>>(qk, amask, (float*)d_out);
}
